// MoELayer_64338610094878
// MI455X (gfx1250) — matrix operand order repaired and checked by reading, not compiled, not run on hardware
//
#include <hip/hip_runtime.h>
#include <math.h>

typedef float v8f  __attribute__((ext_vector_type(8)));
typedef __bf16 v16bf __attribute__((ext_vector_type(16)));

#define T_TOK   8192
#define DMODEL  1024
#define DFF     2048
#define NEXP    16
#define CAP     1280
#define NSLOT   (T_TOK * 2)

// ---------------- bf16 split helpers (Ootomo hi/lo) ----------------
__device__ __forceinline__ unsigned int bf16_rne(float f) {
    unsigned int u = __float_as_uint(f);
    unsigned int r = u + 0x7FFFu + ((u >> 16) & 1u);
    return r >> 16;
}
__device__ __forceinline__ float bf16_val(unsigned int b) {
    return __uint_as_float(b << 16);
}
__device__ __forceinline__ void split2(float v, unsigned int& hi, unsigned int& lo) {
    unsigned int h = bf16_rne(v);
    float r = v - bf16_val(h);
    hi = h;
    lo = bf16_rne(r);
}

union FragU {
    unsigned int u[8];
    uint4 q[2];
    v16bf v;
};

// ---------------- zero output ----------------
__global__ void moe_zero_kernel(float* __restrict__ out, int n) {
    int i = blockIdx.x * 256 + threadIdx.x;
    if (i < n) out[i] = 0.0f;
}

// ---------------- gating: sigmoid + top2 ----------------
__global__ __launch_bounds__(256) void moe_gate_kernel(
    const float* __restrict__ x, const float* __restrict__ gw,
    int* __restrict__ sel, float* __restrict__ wts)
{
    int wave = threadIdx.x >> 5;
    int lane = threadIdx.x & 31;
    int t = blockIdx.x * 8 + wave;
    if (t >= T_TOK) return;

    float acc[NEXP];
#pragma unroll
    for (int e = 0; e < NEXP; ++e) acc[e] = 0.0f;

    const float* xr = x + (size_t)t * DMODEL;
    for (int i = lane; i < DMODEL; i += 32) {
        float xv = xr[i];
#pragma unroll
        for (int e = 0; e < NEXP; ++e) acc[e] += xv * gw[e * DMODEL + i];
    }
#pragma unroll
    for (int e = 0; e < NEXP; ++e) {
        float v = acc[e];
#pragma unroll
        for (int off = 16; off > 0; off >>= 1) v += __shfl_xor(v, off, 32);
        acc[e] = v;
    }
    if (lane == 0) {
        float p[NEXP];
#pragma unroll
        for (int e = 0; e < NEXP; ++e) p[e] = 1.0f / (1.0f + expf(-acc[e]));
        int b0 = 0; float v0 = p[0];
#pragma unroll
        for (int e = 1; e < NEXP; ++e) if (p[e] > v0) { v0 = p[e]; b0 = e; }
        int b1 = -1; float v1 = -1.0f;
#pragma unroll
        for (int e = 0; e < NEXP; ++e) if (e != b0 && p[e] > v1) { v1 = p[e]; b1 = e; }
        float s = v0 + v1 + 1e-6f;
        sel[t * 2 + 0] = b0; sel[t * 2 + 1] = b1;
        wts[t * 2 + 0] = v0 / s; wts[t * 2 + 1] = v1 / s;
    }
}

// ---------------- per-block expert histogram ----------------
__global__ void moe_hist_kernel(const int* __restrict__ sel, int* __restrict__ hist) {
    __shared__ int lh[NEXP];
    int tid = threadIdx.x;
    if (tid < NEXP) lh[tid] = 0;
    __syncthreads();
    int s = blockIdx.x * 256 + tid;
    int e = sel[s];
    atomicAdd(&lh[e], 1);
    __syncthreads();
    if (tid < NEXP) hist[blockIdx.x * NEXP + tid] = lh[tid];
}

// ---------------- exclusive scan over 64 blocks per expert ----------------
__global__ void moe_scan_kernel(const int* __restrict__ hist, int* __restrict__ base) {
    int e = threadIdx.x;
    if (e < NEXP) {
        int run = 0;
        for (int b = 0; b < 64; ++b) {
            base[b * NEXP + e] = run;
            run += hist[b * NEXP + e];
        }
    }
}

// ---------------- init inverse map ----------------
__global__ void moe_initinv_kernel(int* __restrict__ inv) {
    int i = blockIdx.x * 256 + threadIdx.x;
    if (i < NEXP * CAP) inv[i] = -1;
}

// ---------------- ranks + scatter inverse map ----------------
__global__ void moe_rank_kernel(const int* __restrict__ sel, const int* __restrict__ base,
                                int* __restrict__ inv) {
    __shared__ int se[256];
    int tid = threadIdx.x;
    int s = blockIdx.x * 256 + tid;
    int e = sel[s];
    se[tid] = e;
    __syncthreads();
    int local = 0;
    for (int j = 0; j < tid; ++j) local += (se[j] == e) ? 1 : 0;
    int rank = base[blockIdx.x * NEXP + e] + local;
    if (rank < CAP) inv[e * CAP + rank] = s;
}

// ---------------- bf16x3 GEMM tiles ----------------
// block tile 64(M) x 128(N), BK=32 (16 packed k2), 8 waves, wave = 32x32 via 2x2 WMMA bf16
#define BM 64
#define BN 128
#define BKD 32
#define PK 20   // padded row length in packed uints (16 used)

// ---------------- GEMM1: gelu( gather(x) @ w1 ) -> h (f32) ----------------
__global__ __launch_bounds__(256) void moe_gemm1_kernel(
    const float* __restrict__ x, const float* __restrict__ w1,
    const int* __restrict__ inv, float* __restrict__ h)
{
    const int e  = blockIdx.z;
    const int m0 = blockIdx.y * BM;
    const int n0 = blockIdx.x * BN;
    const int tid  = threadIdx.x;
    const int wave = tid >> 5;
    const int lane = tid & 31;

    __shared__ unsigned int AsH[BM][PK];
    __shared__ unsigned int AsL[BM][PK];
    __shared__ unsigned int BsH[BN][PK];
    __shared__ unsigned int BsL[BN][PK];
    __shared__ int rowtok[BM];

    if (tid < BM) {
        int s = inv[e * CAP + m0 + tid];
        rowtok[tid] = (s >= 0) ? (s >> 1) : -1;
    }
    __syncthreads();

    const int wm = (wave >> 2) * 32;
    const int wn = (wave & 3) * 32;
    const float* Bglob = w1 + (size_t)e * DMODEL * DFF;

    v8f acc[2][2] = {};

    for (int k0 = 0; k0 < DMODEL; k0 += BKD) {
        // stage A: 64 rows x 32 k, gathered + hi/lo split, packed k-pairs, m-major
        int idx = tid;
#pragma unroll
        for (int it = 0; it < 2; ++it) {
            int r  = idx >> 3;
            int kg = (idx & 7) * 4;
            int tok = rowtok[r];
            float4 v = make_float4(0.f, 0.f, 0.f, 0.f);
            if (tok >= 0)
                v = *(const float4*)(x + (size_t)tok * DMODEL + k0 + kg);
            unsigned int h0,l0,h1,l1,h2,l2,h3,l3;
            split2(v.x, h0, l0); split2(v.y, h1, l1);
            split2(v.z, h2, l2); split2(v.w, h3, l3);
            uint2 hp, lp;
            hp.x = h0 | (h1 << 16); hp.y = h2 | (h3 << 16);
            lp.x = l0 | (l1 << 16); lp.y = l2 | (l3 << 16);
            *(uint2*)&AsH[r][kg >> 1] = hp;
            *(uint2*)&AsL[r][kg >> 1] = lp;
            idx += 256;
        }
        // stage B: 32 k x 128 n, hi/lo split, packed k-pairs, n-major
        int o = tid;
#pragma unroll
        for (int it = 0; it < 8; ++it) {
            int k2 = o >> 7;
            int n  = o & 127;
            float b0 = Bglob[(size_t)(k0 + 2 * k2 + 0) * DFF + n0 + n];
            float b1 = Bglob[(size_t)(k0 + 2 * k2 + 1) * DFF + n0 + n];
            unsigned int h0,l0,h1,l1;
            split2(b0, h0, l0); split2(b1, h1, l1);
            BsH[n][k2] = h0 | (h1 << 16);
            BsL[n][k2] = l0 | (l1 << 16);
            o += 256;
        }
        __syncthreads();

        const int half = lane >> 4;
        const int l16  = lane & 15;
        FragU aH[2], aL[2], bH[2], bL[2];
#pragma unroll
        for (int mt = 0; mt < 2; ++mt) {
            const unsigned int* apH = &AsH[wm + mt * 16 + l16][0];
            const unsigned int* apL = &AsL[wm + mt * 16 + l16][0];
            aH[mt].q[0] = *(const uint4*)&apH[4 * half];
            aH[mt].q[1] = *(const uint4*)&apH[8 + 4 * half];
            aL[mt].q[0] = *(const uint4*)&apL[4 * half];
            aL[mt].q[1] = *(const uint4*)&apL[8 + 4 * half];
        }
#pragma unroll
        for (int nt = 0; nt < 2; ++nt) {
            const unsigned int* bpH = &BsH[wn + nt * 16 + l16][0];
            const unsigned int* bpL = &BsL[wn + nt * 16 + l16][0];
            bH[nt].q[0] = *(const uint4*)&bpH[4 * half];
            bH[nt].q[1] = *(const uint4*)&bpH[8 + 4 * half];
            bL[nt].q[0] = *(const uint4*)&bpL[4 * half];
            bL[nt].q[1] = *(const uint4*)&bpL[8 + 4 * half];
        }
#pragma unroll
        for (int mt = 0; mt < 2; ++mt)
#pragma unroll
            for (int nt = 0; nt < 2; ++nt) {
                acc[mt][nt] = __builtin_amdgcn_wmma_f32_16x16x32_bf16(
                    false, aH[mt].v, false, bH[nt].v, (short)0, acc[mt][nt], false, false);
                acc[mt][nt] = __builtin_amdgcn_wmma_f32_16x16x32_bf16(
                    false, aH[mt].v, false, bL[nt].v, (short)0, acc[mt][nt], false, false);
                acc[mt][nt] = __builtin_amdgcn_wmma_f32_16x16x32_bf16(
                    false, aL[mt].v, false, bH[nt].v, (short)0, acc[mt][nt], false, false);
            }
        __syncthreads();
    }

    // epilogue: exact gelu, store h (f32)
    float* Hglob = h + (size_t)e * CAP * DFF;
    const int half = lane >> 4;
    const int l16  = lane & 15;
#pragma unroll
    for (int mt = 0; mt < 2; ++mt)
#pragma unroll
        for (int nt = 0; nt < 2; ++nt)
#pragma unroll
            for (int i = 0; i < 8; ++i) {
                int m = m0 + wm + mt * 16 + i + half * 8;
                int n = n0 + wn + nt * 16 + l16;
                float v = acc[mt][nt][i];
                float g = 0.5f * v * (1.0f + erff(v * 0.70710678118654752f));
                Hglob[(size_t)m * DFF + n] = g;
            }
}

// ---------------- GEMM2: h @ w2, weighted atomic scatter -> out ----------------
__global__ __launch_bounds__(256) void moe_gemm2_kernel(
    const float* __restrict__ h, const float* __restrict__ w2,
    const int* __restrict__ inv, const float* __restrict__ wts,
    float* __restrict__ out)
{
    const int e  = blockIdx.z;
    const int m0 = blockIdx.y * BM;
    const int n0 = blockIdx.x * BN;
    const int tid  = threadIdx.x;
    const int wave = tid >> 5;
    const int lane = tid & 31;

    __shared__ unsigned int AsH[BM][PK];
    __shared__ unsigned int AsL[BM][PK];
    __shared__ unsigned int BsH[BN][PK];
    __shared__ unsigned int BsL[BN][PK];
    __shared__ int   rows[BM];
    __shared__ float roww[BM];

    if (tid < BM) {
        int s = inv[e * CAP + m0 + tid];
        rows[tid] = s;
        roww[tid] = (s >= 0) ? wts[s] : 0.0f;
    }
    __syncthreads();

    const int wm = (wave >> 2) * 32;
    const int wn = (wave & 3) * 32;
    const float* Aglob = h  + (size_t)e * CAP * DFF;
    const float* Bglob = w2 + (size_t)e * DFF * DMODEL;

    v8f acc[2][2] = {};

    for (int k0 = 0; k0 < DFF; k0 += BKD) {
        int idx = tid;
#pragma unroll
        for (int it = 0; it < 2; ++it) {
            int r  = idx >> 3;
            int kg = (idx & 7) * 4;
            float4 v = *(const float4*)(Aglob + (size_t)(m0 + r) * DFF + k0 + kg);
            unsigned int h0,l0,h1,l1,h2,l2,h3,l3;
            split2(v.x, h0, l0); split2(v.y, h1, l1);
            split2(v.z, h2, l2); split2(v.w, h3, l3);
            uint2 hp, lp;
            hp.x = h0 | (h1 << 16); hp.y = h2 | (h3 << 16);
            lp.x = l0 | (l1 << 16); lp.y = l2 | (l3 << 16);
            *(uint2*)&AsH[r][kg >> 1] = hp;
            *(uint2*)&AsL[r][kg >> 1] = lp;
            idx += 256;
        }
        int o = tid;
#pragma unroll
        for (int it = 0; it < 8; ++it) {
            int k2 = o >> 7;
            int n  = o & 127;
            float b0 = Bglob[(size_t)(k0 + 2 * k2 + 0) * DMODEL + n0 + n];
            float b1 = Bglob[(size_t)(k0 + 2 * k2 + 1) * DMODEL + n0 + n];
            unsigned int h0,l0,h1,l1;
            split2(b0, h0, l0); split2(b1, h1, l1);
            BsH[n][k2] = h0 | (h1 << 16);
            BsL[n][k2] = l0 | (l1 << 16);
            o += 256;
        }
        __syncthreads();

        const int half = lane >> 4;
        const int l16  = lane & 15;
        FragU aH[2], aL[2], bH[2], bL[2];
#pragma unroll
        for (int mt = 0; mt < 2; ++mt) {
            const unsigned int* apH = &AsH[wm + mt * 16 + l16][0];
            const unsigned int* apL = &AsL[wm + mt * 16 + l16][0];
            aH[mt].q[0] = *(const uint4*)&apH[4 * half];
            aH[mt].q[1] = *(const uint4*)&apH[8 + 4 * half];
            aL[mt].q[0] = *(const uint4*)&apL[4 * half];
            aL[mt].q[1] = *(const uint4*)&apL[8 + 4 * half];
        }
#pragma unroll
        for (int nt = 0; nt < 2; ++nt) {
            const unsigned int* bpH = &BsH[wn + nt * 16 + l16][0];
            const unsigned int* bpL = &BsL[wn + nt * 16 + l16][0];
            bH[nt].q[0] = *(const uint4*)&bpH[4 * half];
            bH[nt].q[1] = *(const uint4*)&bpH[8 + 4 * half];
            bL[nt].q[0] = *(const uint4*)&bpL[4 * half];
            bL[nt].q[1] = *(const uint4*)&bpL[8 + 4 * half];
        }
#pragma unroll
        for (int mt = 0; mt < 2; ++mt)
#pragma unroll
            for (int nt = 0; nt < 2; ++nt) {
                acc[mt][nt] = __builtin_amdgcn_wmma_f32_16x16x32_bf16(
                    false, aH[mt].v, false, bH[nt].v, (short)0, acc[mt][nt], false, false);
                acc[mt][nt] = __builtin_amdgcn_wmma_f32_16x16x32_bf16(
                    false, aH[mt].v, false, bL[nt].v, (short)0, acc[mt][nt], false, false);
                acc[mt][nt] = __builtin_amdgcn_wmma_f32_16x16x32_bf16(
                    false, aL[mt].v, false, bH[nt].v, (short)0, acc[mt][nt], false, false);
            }
        __syncthreads();
    }

    // epilogue: out[token] += weight * val (<=2 contributions per element -> deterministic)
    const int half = lane >> 4;
    const int l16  = lane & 15;
#pragma unroll
    for (int mt = 0; mt < 2; ++mt)
#pragma unroll
        for (int nt = 0; nt < 2; ++nt)
#pragma unroll
            for (int i = 0; i < 8; ++i) {
                int mloc = wm + mt * 16 + i + half * 8;
                int n = n0 + wn + nt * 16 + l16;
                int s = rows[mloc];
                if (s >= 0) {
                    int tok = s >> 1;
                    atomicAdd(&out[(size_t)tok * DMODEL + n], roww[mloc] * acc[mt][nt][i]);
                }
            }
}

// ---------------- launch ----------------
extern "C" void kernel_launch(void* const* d_in, const int* in_sizes, int n_in,
                              void* d_out, int out_size, void* d_ws, size_t ws_size,
                              hipStream_t stream) {
    const float* x      = (const float*)d_in[0];
    const float* gate_w = (const float*)d_in[1];
    const float* w1     = (const float*)d_in[2];
    const float* w2     = (const float*)d_in[3];
    float* out = (float*)d_out;

    char* ws = (char*)d_ws;
    int*   sel  = (int*)(ws + 0);                  // 16384 ints
    float* wts  = (float*)(ws + 65536);            // 16384 floats
    int*   hist = (int*)(ws + 131072);             // 64*16
    int*   base = (int*)(ws + 135168);             // 64*16
    int*   inv  = (int*)(ws + 139264);             // 16*1280
    float* h    = (float*)(ws + 262144);           // 16*1280*2048 floats (160 MB)

    const int out_elems = T_TOK * DMODEL;
    moe_zero_kernel<<<(out_elems + 255) / 256, 256, 0, stream>>>(out, out_elems);
    moe_gate_kernel<<<T_TOK / 8, 256, 0, stream>>>(x, gate_w, sel, wts);
    moe_hist_kernel<<<NSLOT / 256, 256, 0, stream>>>(sel, hist);
    moe_scan_kernel<<<1, 32, 0, stream>>>(hist, base);
    moe_initinv_kernel<<<(NEXP * CAP + 255) / 256, 256, 0, stream>>>(inv);
    moe_rank_kernel<<<NSLOT / 256, 256, 0, stream>>>(sel, base, inv);

    dim3 g1(DFF / BN, CAP / BM, NEXP);
    moe_gemm1_kernel<<<g1, 256, 0, stream>>>(x, w1, inv, h);
    dim3 g2(DMODEL / BN, CAP / BM, NEXP);
    moe_gemm2_kernel<<<g2, 256, 0, stream>>>(h, w2, inv, wts, out);
}
